// QuantizedLinear_29317446762953
// MI455X (gfx1250) — matrix operand order repaired and checked by reading, not compiled, not run on hardware
//
#include <hip/hip_runtime.h>

typedef __attribute__((ext_vector_type(16))) _Float16 v16h;
typedef __attribute__((ext_vector_type(8)))  _Float16 v8h;
typedef __attribute__((ext_vector_type(8)))  float    v8f;
typedef __attribute__((ext_vector_type(4)))  int      v4i;
typedef __attribute__((ext_vector_type(4)))  float    v4f;

#define TOKENS 8192
#define OUTF   4096
#define INF    4096
#define NGRP   32

#define BM 128
#define BN 128
#define BK 64
#define KPAD 72            // padded LDS row stride in halfs (144B): conflict-free frags
#define TILE_HALFS (BM * KPAD)          // one tile (A or B), in halfs
#define BUF_BYTES  (TILE_HALFS * 2)     // one tile, in bytes (A/B same: BM==BN)
#define NITER      (INF / BK)           // 64 k-iterations

// ---------------------------------------------------------------------------
// CDNA5 async global->LDS copy (VGLOBAL GVS form, tracked by ASYNCcnt).
//   lds : LDS byte address (low 32 bits of generic shared pointer)
//   voff: per-lane signed 32-bit byte offset from sbase
//   sbase: uniform 64-bit global base (SGPR pair)
// ---------------------------------------------------------------------------
__device__ __forceinline__ void async_copy_b128(uint32_t lds, uint32_t voff,
                                                const _Float16* sbase) {
    asm volatile("global_load_async_to_lds_b128 %0, %1, %2"
                 :: "v"(lds), "v"(voff), "s"(sbase)
                 : "memory");
}

__device__ __forceinline__ void wait_async_zero() {
    asm volatile("s_wait_asynccnt 0x0" ::: "memory");
}

// ---------------------------------------------------------------------------
// Kernel 1: dequantize int4-in-int32 weights -> f16, row-major [OUT][IN]
// ---------------------------------------------------------------------------
__global__ __launch_bounds__(256) void dequant_w_f16(
    const int* __restrict__ qw, const float* __restrict__ scales,
    const float* __restrict__ zeros, _Float16* __restrict__ wh)
{
    const int idx = blockIdx.x * 256 + threadIdx.x;   // one thread per 8 elems
    const int e   = idx * 8;
    const int o   = e >> 12;            // / IN
    const int i   = e & (INF - 1);
    const int g   = i >> 7;             // / GROUP(128); 8 elems never cross group
    const float s = scales[o * NGRP + g];
    const float z = zeros [o * NGRP + g];
    const v4i q0 = *reinterpret_cast<const v4i*>(qw + e);
    const v4i q1 = *reinterpret_cast<const v4i*>(qw + e + 4);
    v8h h;
    h[0] = (_Float16)(((float)q0[0] - z) * s);
    h[1] = (_Float16)(((float)q0[1] - z) * s);
    h[2] = (_Float16)(((float)q0[2] - z) * s);
    h[3] = (_Float16)(((float)q0[3] - z) * s);
    h[4] = (_Float16)(((float)q1[0] - z) * s);
    h[5] = (_Float16)(((float)q1[1] - z) * s);
    h[6] = (_Float16)(((float)q1[2] - z) * s);
    h[7] = (_Float16)(((float)q1[3] - z) * s);
    *reinterpret_cast<v8h*>(wh + e) = h;
}

// ---------------------------------------------------------------------------
// Kernel 2: convert input f32 -> f16, row-major [TOKENS][IN]
// ---------------------------------------------------------------------------
__global__ __launch_bounds__(256) void cvt_x_f16(
    const float* __restrict__ x, _Float16* __restrict__ xh)
{
    const int idx = blockIdx.x * 256 + threadIdx.x;
    const int e   = idx * 8;
    const v4f a = *reinterpret_cast<const v4f*>(x + e);
    const v4f b = *reinterpret_cast<const v4f*>(x + e + 4);
    v8h h;
    h[0] = (_Float16)a[0]; h[1] = (_Float16)a[1];
    h[2] = (_Float16)a[2]; h[3] = (_Float16)a[3];
    h[4] = (_Float16)b[0]; h[5] = (_Float16)b[1];
    h[6] = (_Float16)b[2]; h[7] = (_Float16)b[3];
    *reinterpret_cast<v8h*>(xh + e) = h;
}

// ---------------------------------------------------------------------------
// Kernel 3: WMMA f16 GEMM  out = Xh * Wh^T + bias
//   grid (OUT/BN, TOKENS/BM), 256 threads = 8 waves (wave32)
//   wave grid 2(M) x 4(N); each wave: 64x32 tile = 4 mtiles x 2 ntiles
//   Double-buffered LDS fed by async global->LDS copies (ASYNCcnt pipeline).
// ---------------------------------------------------------------------------
__global__ __launch_bounds__(256) void gemm_wmma_f16(
    const _Float16* __restrict__ Xh, const _Float16* __restrict__ Wh,
    const float* __restrict__ bias, float* __restrict__ out)
{
    extern __shared__ _Float16 smem[];
    _Float16* sA = smem;                      // [2][BM*KPAD]
    _Float16* sB = smem + 2 * TILE_HALFS;     // [2][BN*KPAD]

    const int tid   = threadIdx.x;
    const int lane  = tid & 31;
    const int wave  = tid >> 5;
    const int waveM = wave & 1;    // 0..1
    const int waveN = wave >> 1;   // 0..3
    const int l16   = lane & 15;
    const int lhi   = lane >> 4;   // 0 or 1

    const int blockN = blockIdx.x * BN;
    const int blockM = blockIdx.y * BM;

    // Per-thread copy coordinates: 1024 16B segments per tile, 4 per thread.
    uint32_t ldsA0[4], ldsB0[4];   // LDS byte addresses (buffer 0)
    uint32_t voffA[4], voffB[4];   // global byte offsets at k=0
#pragma unroll
    for (int i = 0; i < 4; i++) {
        const int s   = tid + 256 * i;
        const int row = s >> 3;
        const int sc8 = (s & 7) * 8;
        ldsA0[i] = (uint32_t)(uintptr_t)&sA[row * KPAD + sc8];
        ldsB0[i] = (uint32_t)(uintptr_t)&sB[row * KPAD + sc8];
        voffA[i] = (uint32_t)(((blockM + row) * INF + sc8) * 2);
        voffB[i] = (uint32_t)(((blockN + row) * INF + sc8) * 2);
    }

    auto issueCopies = [&](int kb, int buf) {
        const uint32_t k0b  = (uint32_t)(kb * BK * 2);   // byte advance along K
        const uint32_t boff = (uint32_t)(buf * BUF_BYTES);
#pragma unroll
        for (int i = 0; i < 4; i++) {
            async_copy_b128(ldsA0[i] + boff, voffA[i] + k0b, Xh);
            async_copy_b128(ldsB0[i] + boff, voffB[i] + k0b, Wh);
        }
    };

    v8f acc[4][2] = {};   // zero-initialized accumulators

    issueCopies(0, 0);
    for (int kb = 0; kb < NITER; kb++) {
        const int cur = kb & 1;
        wait_async_zero();    // my 8 async copies for buffer `cur` are done
        __syncthreads();      // all waves' tiles landed; prev compute finished
        if (kb + 1 < NITER) issueCopies(kb + 1, cur ^ 1);  // overlap w/ WMMA

        const _Float16* cA = sA + cur * TILE_HALFS;
        const _Float16* cB = sB + cur * TILE_HALFS;

#pragma unroll
        for (int k32 = 0; k32 < 2; k32++) {
            v16h afr[4];
            v16h bfr[2];
#pragma unroll
            for (int mt = 0; mt < 4; mt++) {
                // A frag (16x32 f16): lane = M%16; lanes0-15 K 0-7 & 16-23,
                // lanes16-31 K 8-15 & 24-31
                const int r    = waveM * 64 + mt * 16 + l16;
                const int koff = k32 * 32 + lhi * 8;
                const v8h lo = *reinterpret_cast<const v8h*>(&cA[r * KPAD + koff]);
                const v8h hi = *reinterpret_cast<const v8h*>(&cA[r * KPAD + koff + 16]);
                afr[mt] = __builtin_shufflevector(lo, hi,
                    0,1,2,3,4,5,6,7,8,9,10,11,12,13,14,15);
            }
#pragma unroll
            for (int nt = 0; nt < 2; nt++) {
                // B frag (32x16 f16): lane = N%16; lanes0-15 K 0-7 & 16-23,
                // lanes16-31 K 8-15 & 24-31 (Wh is [N][K] so K is contiguous)
                const int c    = waveN * 32 + nt * 16 + l16;
                const int koff = k32 * 32 + lhi * 8;
                const v8h lo = *reinterpret_cast<const v8h*>(&cB[c * KPAD + koff]);
                const v8h hi = *reinterpret_cast<const v8h*>(&cB[c * KPAD + koff + 16]);
                bfr[nt] = __builtin_shufflevector(lo, hi,
                    0,1,2,3,4,5,6,7,8,9,10,11,12,13,14,15);
            }
#pragma unroll
            for (int mt = 0; mt < 4; mt++)
#pragma unroll
                for (int nt = 0; nt < 2; nt++)
                    acc[mt][nt] = __builtin_amdgcn_wmma_f32_16x16x32_f16(
                        false, afr[mt], false, bfr[nt],
                        (short)0, acc[mt][nt], false, false);
        }
    }

    // Epilogue: C/D layout -> VGPR r: lanes0-15 M=r, lanes16-31 M=8+r; N=lane%16
#pragma unroll
    for (int nt = 0; nt < 2; nt++) {
        const int o  = blockN + waveN * 32 + nt * 16 + l16;
        const float bv = bias[o];
#pragma unroll
        for (int mt = 0; mt < 4; mt++) {
            const int tbase = blockM + waveM * 64 + mt * 16 + lhi * 8;
#pragma unroll
            for (int r = 0; r < 8; r++) {
                out[(size_t)(tbase + r) * OUTF + o] = acc[mt][nt][r] + bv;
            }
        }
    }
}

// ---------------------------------------------------------------------------
extern "C" void kernel_launch(void* const* d_in, const int* in_sizes, int n_in,
                              void* d_out, int out_size, void* d_ws, size_t ws_size,
                              hipStream_t stream) {
    const float* x     = (const float*)d_in[0];   // [8192,4096] f32
    const int*   qw    = (const int*)  d_in[1];   // [4096,4096] int32 (0..15)
    const float* sc    = (const float*)d_in[2];   // [4096,32]
    const float* zp    = (const float*)d_in[3];   // [4096,32]
    const float* bias  = (const float*)d_in[4];   // [4096]
    float*       out   = (float*)d_out;           // [8192,4096] f32

    _Float16* wh = (_Float16*)d_ws;                                   // 33.5 MB
    _Float16* xh = (_Float16*)((char*)d_ws + (size_t)OUTF * INF * 2); // 67 MB

    dequant_w_f16<<<(OUTF * INF / 8) / 256, 256, 0, stream>>>(qw, sc, zp, wh);
    cvt_x_f16<<<(TOKENS * INF / 8) / 256, 256, 0, stream>>>(x, xh);

    const size_t shmem = 4 * TILE_HALFS * sizeof(_Float16);  // 73728 B (<320KB/WGP)
    dim3 grid(OUTF / BN, TOKENS / BM);   // (32, 64)
    gemm_wmma_f16<<<grid, 256, shmem, stream>>>(xh, wh, bias, out);
}
